// GraphAutoencoder_31035433681218
// MI455X (gfx1250) — matrix operand order repaired and checked by reading, not compiled, not run on hardware
//
#include <hip/hip_runtime.h>
#include <stdint.h>
#include <stddef.h>

// ---------------------------------------------------------------------------
// GraphAutoencoder on MI455X (gfx1250, wave32, WMMA bf16 16x16x32, f32 accum)
//
// ws layout (bytes, ~260 MB total):
//   h    : [50000][512] bf16   (encoder output, padded cols 500..511 = 0)
//   h2   : [50000][512] bf16   (conv output)
//   hout : [50000][512] bf16   (agg + h @ lin^T)
//   agg  : [50000][512] f32    (scatter accumulator, memset 0 each call)
//   wencP: [512][1024]  bf16   (wenc rows, zero pad rows 500..511)
//   convT: [512][512]   bf16   (conv_weight^T, zero padded)
//   linP : [512][512]   bf16   (lin_weight rows, zero padded)
//   wdecP: [1024][512]  bf16   (wdec rows, K zero-padded 500..511)
// ---------------------------------------------------------------------------

#define NN      50000
#define NE      800000
#define INSZ    1024
#define EMB     500
#define EMBP    512

#define BM      128
#define BN      128
#define BK      32
#define LDT     40      // LDS row stride in bf16 elements (32 data + 8 pad) => 80B, conflict-free

typedef __attribute__((ext_vector_type(16))) __bf16 v16bf;
typedef __attribute__((ext_vector_type(8)))  float  v8f;

__device__ __forceinline__ unsigned short f2bf(float f) {
  union { float f; unsigned u; } x; x.f = f;
  unsigned r = x.u + 0x7FFFu + ((x.u >> 16) & 1u);   // round-nearest-even
  return (unsigned short)(r >> 16);
}
__device__ __forceinline__ float bf2f(unsigned short b) {
  union { unsigned u; float f; } x; x.u = ((unsigned)b) << 16;
  return x.f;
}

// --- weight pack: D[r][c] = (r<srcR && c<srcC) ? (T ? S[c*sld+r] : S[r*sld+c]) : 0 ---
__global__ void pack_weight(const float* __restrict__ S, unsigned short* __restrict__ D,
                            int R, int C, int srcR, int srcC, int sld, int T) {
  int idx = blockIdx.x * blockDim.x + threadIdx.x;
  if (idx >= R * C) return;
  int r = idx / C, c = idx % C;
  float v = 0.f;
  if (r < srcR && c < srcC) v = T ? S[(size_t)c * sld + r] : S[(size_t)r * sld + c];
  D[(size_t)r * C + c] = f2bf(v);
}

// --- edge scatter: agg[dst] += h2[src] * w  (f32 atomics, agg resident in 192MB L2) ---
__global__ void scatter_edges(const int* __restrict__ src, const int* __restrict__ dst,
                              const float* __restrict__ ew,
                              const unsigned short* __restrict__ h2,
                              float* __restrict__ agg) {
  int e = blockIdx.x;
  int s = src[e], d = dst[e];
  float w = ew[e];
  const unsigned short* hr = h2 + (size_t)s * EMBP;
  float* ar = agg + (size_t)d * EMBP;
  for (int c = threadIdx.x; c < EMB; c += blockDim.x)
    atomicAdd(ar + c, bf2f(hr[c]) * w);
}

// --- epilogue selector ---
enum { EPI_ENC = 0, EPI_PLAIN = 1, EPI_ADD = 2, EPI_DEC = 3 };

// Tiled GEMM: C[M,N] = A[M,K] x B^T where B is packed row-major [N][K] bf16.
// 256 threads = 8 waves; wave tile 32(M) x 64(N) = 2x4 WMMA frags; double-buffered LDS.
template<bool AF32, int EPI>
__global__ __launch_bounds__(256)
void gemm_wmma(const void* __restrict__ Ap, const unsigned short* __restrict__ Bp,
               void* __restrict__ Cp, const float* __restrict__ bias,
               const float* __restrict__ addp, int M, int K, int ldc, int nmask)
{
  __shared__ unsigned short sA[2][BM * LDT];
  __shared__ unsigned short sB[2][BN * LDT];

  const int tid  = threadIdx.x;
  const int lane = tid & 31;
  const int w    = tid >> 5;
  const int lm   = lane & 15;
  const int lh   = lane >> 4;
  const int waveM = (w >> 1) * 32;    // 4 waves along M
  const int waveN = (w & 1) * 64;     // 2 waves along N
  const int bMo = blockIdx.x * BM;
  const int bNo = blockIdx.y * BN;

  const float*          Af = (const float*)Ap;
  const unsigned short* Ab = (const unsigned short*)Ap;

  v8f acc[2][4];
  #pragma unroll
  for (int i = 0; i < 2; ++i)
    #pragma unroll
    for (int j = 0; j < 4; ++j)
      #pragma unroll
      for (int k = 0; k < 8; ++k) acc[i][j][k] = 0.f;

  auto stageA = [&](int buf, int k0) {
    if (AF32) {
      #pragma unroll
      for (int i = 0; i < 4; ++i) {                 // 1024 float4 / 256 thr
        int v = i * 256 + tid;
        int row = v >> 3;                           // 8 float4 per 32-col row
        int c4  = (v & 7) << 2;
        int gr = bMo + row;
        float4 f = make_float4(0.f, 0.f, 0.f, 0.f);
        if (gr < M) f = *(const float4*)(Af + (size_t)gr * K + k0 + c4);
        unsigned short* d = &sA[buf][row * LDT + c4];
        d[0] = f2bf(f.x); d[1] = f2bf(f.y); d[2] = f2bf(f.z); d[3] = f2bf(f.w);
      }
    } else {
      #pragma unroll
      for (int i = 0; i < 2; ++i) {                 // 512 x 16B / 256 thr
        int v = i * 256 + tid;
        int row = v >> 2;                           // 4 x 16B per row
        int c8  = (v & 3) << 3;
        int gr = bMo + row;
        uint4 q = make_uint4(0u, 0u, 0u, 0u);
        if (gr < M) q = *(const uint4*)(Ab + (size_t)gr * K + k0 + c8);
        *(uint4*)(&sA[buf][row * LDT + c8]) = q;
      }
    }
  };

  auto stageB = [&](int buf, int k0) {
    #pragma unroll
    for (int i = 0; i < 2; ++i) {
      int v = i * 256 + tid;
      int row = v >> 2;
      int c8  = (v & 3) << 3;
      uint4 q = *(const uint4*)(Bp + (size_t)(bNo + row) * K + k0 + c8);
      *(uint4*)(&sB[buf][row * LDT + c8]) = q;
    }
  };

  auto compute = [&](int buf) {
    union { v16bf v; uint4 q[2]; } a[2], b[4];
    // A frag (ISA 16-bit A 16x32): lane half h: VGPR0-3 -> K=8h..8h+7, VGPR4-7 -> K=16+8h..
    #pragma unroll
    for (int mi = 0; mi < 2; ++mi) {
      const char* pc = (const char*)&sA[buf][(waveM + mi * 16 + lm) * LDT] + 16 * lh;
      a[mi].q[0] = *(const uint4*)(pc);
      a[mi].q[1] = *(const uint4*)(pc + 32);
    }
    // B frag: lane col = lm, same K order as A: VGPR0-3 -> K=8h..8h+7, VGPR4-7 -> K=16+8h..
    #pragma unroll
    for (int nj = 0; nj < 4; ++nj) {
      const char* pc = (const char*)&sB[buf][(waveN + nj * 16 + lm) * LDT] + 16 * lh;
      b[nj].q[0] = *(const uint4*)(pc);
      b[nj].q[1] = *(const uint4*)(pc + 32);
    }
    #pragma unroll
    for (int mi = 0; mi < 2; ++mi)
      #pragma unroll
      for (int nj = 0; nj < 4; ++nj)
        acc[mi][nj] = __builtin_amdgcn_wmma_f32_16x16x32_bf16(
            false, a[mi].v, false, b[nj].v, (short)0, acc[mi][nj], false, false);
  };

  stageA(0, 0); stageB(0, 0);
  __syncthreads();
  const int nk = K / BK;
  for (int kt = 0; kt < nk; ++kt) {
    int cur = kt & 1;
    if (kt + 1 < nk) { stageA(cur ^ 1, (kt + 1) * BK); stageB(cur ^ 1, (kt + 1) * BK); }
    compute(cur);
    __syncthreads();
  }

  // Epilogue. C/D layout: lane col = lm, row = base + 8*lh + v (v = vgpr index)
  float*          Cf = (float*)Cp;
  unsigned short* Cb = (unsigned short*)Cp;
  #pragma unroll
  for (int mi = 0; mi < 2; ++mi) {
    #pragma unroll
    for (int nj = 0; nj < 4; ++nj) {
      int gcol  = bNo + waveN + nj * 16 + lm;
      int rbase = bMo + waveM + mi * 16 + lh * 8;
      #pragma unroll
      for (int v = 0; v < 8; ++v) {
        int grow = rbase + v;
        if (grow >= M) continue;
        float val = acc[mi][nj][v];
        size_t off = (size_t)grow * ldc + gcol;
        if (EPI == EPI_ENC) {
          float s = 0.f;
          if (gcol < nmask) s = 1.f / (1.f + expf(-(val + bias[gcol])));
          Cb[off] = f2bf(s);                      // pad cols forced to exact 0
        } else if (EPI == EPI_PLAIN) {
          Cb[off] = f2bf(val);
        } else if (EPI == EPI_ADD) {
          Cb[off] = f2bf(val + addp[off]);
        } else {                                  // EPI_DEC
          Cf[off] = val + bias[gcol];
        }
      }
    }
  }
}

extern "C" void kernel_launch(void* const* d_in, const int* in_sizes, int n_in,
                              void* d_out, int out_size, void* d_ws, size_t ws_size,
                              hipStream_t stream) {
  (void)in_sizes; (void)n_in; (void)out_size; (void)ws_size;
  const float* x     = (const float*)d_in[0];
  const int*   eidx  = (const int*)  d_in[1];   // [2][NE]
  const float* ew    = (const float*)d_in[2];
  const float* wenc  = (const float*)d_in[3];   // [500][1024]
  const float* benc  = (const float*)d_in[4];   // [500]
  const float* wdec  = (const float*)d_in[5];   // [1024][500]
  const float* bdec  = (const float*)d_in[6];   // [1024]
  const float* convw = (const float*)d_in[7];   // [500][500]
  const float* linw  = (const float*)d_in[8];   // [500][500]
  float* out = (float*)d_out;

  char* ws = (char*)d_ws;
  size_t off = 0;
  auto alloc = [&](size_t bytes) -> void* {
    void* p = ws + off; off += (bytes + 255) & ~(size_t)255; return p;
  };
  unsigned short* h     = (unsigned short*)alloc((size_t)NN * EMBP * 2);
  unsigned short* h2    = (unsigned short*)alloc((size_t)NN * EMBP * 2);
  unsigned short* hout  = (unsigned short*)alloc((size_t)NN * EMBP * 2);
  float*          agg   = (float*)         alloc((size_t)NN * EMBP * 4);
  unsigned short* wencP = (unsigned short*)alloc((size_t)EMBP * INSZ * 2);
  unsigned short* convT = (unsigned short*)alloc((size_t)EMBP * EMBP * 2);
  unsigned short* linP  = (unsigned short*)alloc((size_t)EMBP * EMBP * 2);
  unsigned short* wdecP = (unsigned short*)alloc((size_t)INSZ * EMBP * 2);

  hipMemsetAsync(agg, 0, (size_t)NN * EMBP * 4, stream);

  const int pt = 256;
  pack_weight<<<(EMBP * INSZ + pt - 1) / pt, pt, 0, stream>>>(wenc,  wencP, EMBP, INSZ, EMB,  INSZ, INSZ, 0);
  pack_weight<<<(EMBP * EMBP + pt - 1) / pt, pt, 0, stream>>>(convw, convT, EMBP, EMBP, EMB,  EMB,  EMB,  1);
  pack_weight<<<(EMBP * EMBP + pt - 1) / pt, pt, 0, stream>>>(linw,  linP,  EMBP, EMBP, EMB,  EMB,  EMB,  0);
  pack_weight<<<(INSZ * EMBP + pt - 1) / pt, pt, 0, stream>>>(wdec,  wdecP, INSZ, EMBP, INSZ, EMB,  EMB,  0);

  dim3 blk(256);
  dim3 gE((NN + BM - 1) / BM, EMBP / BN);   // 391 x 4
  dim3 gD((NN + BM - 1) / BM, INSZ / BN);   // 391 x 8

  // 1) h = sigmoid(x @ wenc^T + benc)          (A = f32 x, convert-on-stage)
  gemm_wmma<true,  EPI_ENC  ><<<gE, blk, 0, stream>>>(x,    wencP, h,    benc, nullptr, NN, INSZ, EMBP, EMB);
  // 2) h2 = h @ conv_weight
  gemm_wmma<false, EPI_PLAIN><<<gE, blk, 0, stream>>>(h,    convT, h2,   nullptr, nullptr, NN, EMBP, EMBP, EMB);
  // 3) agg[dst] += h2[src] * w
  scatter_edges<<<NE, 256, 0, stream>>>(eidx, eidx + NE, ew, h2, agg);
  // 4) hout = agg + h @ lin^T
  gemm_wmma<false, EPI_ADD  ><<<gE, blk, 0, stream>>>(h,    linP,  hout, nullptr, agg, NN, EMBP, EMBP, EMB);
  // 5) out = hout @ wdec^T + bdec
  gemm_wmma<false, EPI_DEC  ><<<gD, blk, 0, stream>>>(hout, wdecP, out,  bdec, nullptr, NN, EMBP, INSZ, INSZ);
}
